// SpatialLoss_38886633898191
// MI455X (gfx1250) — matrix operand order repaired and checked by reading, not compiled, not run on hardware
//
#include <hip/hip_runtime.h>
#include <hip/hip_bf16.h>

typedef __attribute__((ext_vector_type(16))) _Float16 v16h;
typedef __attribute__((ext_vector_type(8)))  _Float16 v8h;
typedef __attribute__((ext_vector_type(8)))  float    v8f;

#define BN 4096
#define DN 512
#define KN 8
#define CSPLIT 8                       // column chunks per direction
#define COLS_PER_CHUNK (BN / CSPLIT)   // 512
#define ROWS_PER_BLOCK 128             // 8 waves * 16 rows
#define STAGE_COLS 32                  // columns staged in LDS per step
#define PADH (DN + 8)                  // halfs per staged column: 1040 B
                                       //  -> lane l base bank = 4*l mod 64,
                                       //     conflict-free ds_load_b128
#define LDS_BYTES (2 * STAGE_COLS * PADH * 2)   // 66,560 B ping-pong

// ---------------------------------------------------------------------------
// Kernel 1: convert f32 features -> f16 workspace copies, build inverse perms
// ---------------------------------------------------------------------------
__global__ void prep_kernel(const float* __restrict__ img,
                            const float* __restrict__ txt,
                            const int* __restrict__ imgIds,
                            const int* __restrict__ txtIds,
                            _Float16* __restrict__ imgH,
                            _Float16* __restrict__ txtH,
                            int* __restrict__ invImg,
                            int* __restrict__ invTxt) {
    int idx = blockIdx.x * blockDim.x + threadIdx.x;
    if (idx < BN * DN) {
        imgH[idx] = (_Float16)img[idx];
        txtH[idx] = (_Float16)txt[idx];
    }
    if (idx < BN) {
        // tile ids are permutations of 0..B-1 -> exact inverse lookup
        invImg[imgIds[idx]] = idx;
        invTxt[txtIds[idx]] = idx;
    }
}

// ---------------------------------------------------------------------------
// Kernel 2: streaming online-logsumexp over rows of s * (A @ B^T) via WMMA.
// Block = 8 waves = 128 rows; blockIdx.y = column chunk of 512 columns.
// B columns staged in double-buffered LDS via global_load_async_to_lds_b128
// (loaded once per block, consumed by all 8 waves). A tile held in VGPRs.
// ---------------------------------------------------------------------------
__global__ void __launch_bounds__(256)
lse_kernel(const _Float16* __restrict__ A,     // [BN][DN] row-major f16
           const _Float16* __restrict__ Bm,    // [BN][DN] row-major f16
           const float* __restrict__ scalePtr, // logit_scale (1 elem)
           float* __restrict__ partM,          // [CSPLIT][BN]
           float* __restrict__ partS) {        // [CSPLIT][BN]
    extern __shared__ _Float16 smem[];         // [2][STAGE_COLS][PADH]

    const int lane  = threadIdx.x & 31;
    const int wave  = threadIdx.x >> 5;
    const int half  = lane >> 4;     // 0: lanes 0-15, 1: lanes 16-31
    const int l15   = lane & 15;
    const int rowBase  = blockIdx.x * ROWS_PER_BLOCK + wave * 16;
    const int colChunk = blockIdx.y;
    const int chunkBase = colChunk * COLS_PER_CHUNK;

    const float sEff = fminf(scalePtr[0], 100.0f);  // capped logit scale

    // A operand k order used here (16-bit A 16x32, contiguous, same as B): lane l holds row l&15,
    // K = abase + 0..15, abase = half?16:0
    const int abase = half ? 16 : 0;
    // B operand k order used here (32x16, contiguous, same as A): lane l holds col l&15, K = kbase + 0..15
    const int kbase = half ? 16 : 0;

    // Preload the wave's 16-row A tile into registers (128 VGPRs), reused
    // across all column tiles.
    const _Float16* aRowPtr = A + (size_t)(rowBase + l15) * DN;
    v16h aFrag[16];
#pragma unroll
    for (int t = 0; t < 16; ++t) {
        v8h a0 = *(const v8h*)(aRowPtr + t * 32 + abase);
        v8h a1 = *(const v8h*)(aRowPtr + t * 32 + abase + 8);
#pragma unroll
        for (int e = 0; e < 8; ++e) { aFrag[t][e] = a0[e]; aFrag[t][e + 8] = a1[e]; }
    }

    // Async-stage STAGE_COLS columns of B into LDS buffer q.
    // 32 cols * 1 KB = 2048 16-byte chunks over 256 threads = 8 per thread.
    auto prefetchStage = [&](int q, int ct) {
        _Float16* stage = smem + q * (STAGE_COLS * PADH);
        const int c0 = chunkBase + ct * STAGE_COLS;
        const int tid = threadIdx.x;
#pragma unroll
        for (int j = 0; j < 8; ++j) {
            int chunk = tid + j * 256;          // 16-byte granules
            int colL  = chunk >> 6;             // 64 granules per column
            int kOff  = (chunk & 63) * 8;       // in halfs
            const _Float16* g = Bm + (size_t)(c0 + colL) * DN + kOff;
            unsigned int ldsOff =
                (unsigned int)(unsigned long long)(stage + colL * PADH + kOff);
            asm volatile("global_load_async_to_lds_b128 %0, %1, off"
                         :: "v"(ldsOff), "v"(g) : "memory");
        }
    };

    float m[8], s[8];
#pragma unroll
    for (int r = 0; r < 8; ++r) { m[r] = -3.0e38f; s[r] = 0.0f; }

    prefetchStage(0, 0);

    for (int ct = 0; ct < COLS_PER_CHUNK / STAGE_COLS; ++ct) {
        const int P = ct & 1;
        // my async loads done, then block-wide: stage P full, stage !P free
        asm volatile("s_wait_asynccnt 0" ::: "memory");
        __syncthreads();
        if (ct + 1 < COLS_PER_CHUNK / STAGE_COLS)
            prefetchStage(P ^ 1, ct + 1);       // overlaps with compute below

        const _Float16* st = smem + P * (STAGE_COLS * PADH);
        const _Float16* b0p = st + l15 * PADH + kbase;         // tile cols 0-15
        const _Float16* b1p = st + (16 + l15) * PADH + kbase;  // tile cols 16-31

        v8f c0 = {}, c1 = {};
        v16h b0 = *(const v16h*)(b0p);
        v16h b1 = *(const v16h*)(b1p);
#pragma unroll
        for (int t = 0; t < 16; ++t) {
            v16h p0 = b0, p1 = b1;
            if (t < 15) {                 // LDS prefetch of next K-step
                b0 = *(const v16h*)(b0p + (t + 1) * 32);
                b1 = *(const v16h*)(b1p + (t + 1) * 32);
            }
            c0 = __builtin_amdgcn_wmma_f32_16x16x32_f16(
                     false, aFrag[t], false, p0, (short)0, c0, false, false);
            c1 = __builtin_amdgcn_wmma_f32_16x16x32_f16(
                     false, aFrag[t], false, p1, (short)0, c1, false, false);
        }

        // online softmax update; lane holds Z[rowBase + r + half*8][cols]
#pragma unroll
        for (int r = 0; r < 8; ++r) {
            float z0 = sEff * c0[r];
            float mN = fmaxf(m[r], z0);
            s[r] = s[r] * __expf(m[r] - mN) + __expf(z0 - mN);
            m[r] = mN;
            float z1 = sEff * c1[r];
            float mN1 = fmaxf(m[r], z1);
            s[r] = s[r] * __expf(m[r] - mN1) + __expf(z1 - mN1);
            m[r] = mN1;
        }
        __syncthreads();   // all waves done reading stage P before next write
    }

    // merge (m,s) across the 16 lanes of each half (columns col%16 = l15)
#pragma unroll
    for (int off = 1; off < 16; off <<= 1) {
#pragma unroll
        for (int r = 0; r < 8; ++r) {
            float mo = __shfl_xor(m[r], off, 32);
            float so = __shfl_xor(s[r], off, 32);
            float mN = fmaxf(m[r], mo);
            s[r] = s[r] * __expf(m[r] - mN) + so * __expf(mo - mN);
            m[r] = mN;
        }
    }

    if (l15 == 0) {
#pragma unroll
        for (int r = 0; r < 8; ++r) {
            int row = rowBase + r + half * 8;
            partM[colChunk * BN + row] = m[r];
            partS[colChunk * BN + row] = s[r];
        }
    }
}

// ---------------------------------------------------------------------------
// Kernel 3: per-row loss. One wave per row: merge chunk partials -> lse,
// gather <=9 dot products (diagonal + neighbors via inverse permutation).
// ---------------------------------------------------------------------------
__global__ void __launch_bounds__(256)
row_loss_kernel(const _Float16* __restrict__ AH,
                const _Float16* __restrict__ BH,
                const float* __restrict__ scalePtr,
                const float* __restrict__ alphas,   // [BN][KN]
                const int* __restrict__ nbIds,      // [BN][KN]
                const int* __restrict__ inv,        // [BN]
                const float* __restrict__ partM,
                const float* __restrict__ partS,
                float* __restrict__ rowLoss) {
    const int lane = threadIdx.x & 31;
    const int wave = threadIdx.x >> 5;
    const int row  = blockIdx.x * 8 + wave;
    if (row >= BN) return;

    const float sEff = fminf(scalePtr[0], 100.0f);

    // merge the CSPLIT online-softmax partials (redundant across lanes, cheap)
    float m = -3.0e38f, ssum = 0.0f;
#pragma unroll
    for (int c = 0; c < CSPLIT; ++c) {
        float mc = partM[c * BN + row];
        float sc = partS[c * BN + row];
        float mN = fmaxf(m, mc);
        ssum = ssum * __expf(m - mN) + sc * __expf(mc - mN);
        m = mN;
    }
    const float lse = m + __logf(ssum);

    // each lane covers 16 contiguous elements of the 512-dim row
    v16h a = *(const v16h*)(AH + (size_t)row * DN + lane * 16);

    auto dotWith = [&](int col) -> float {
        v16h b = *(const v16h*)(BH + (size_t)col * DN + lane * 16);
        float acc = 0.0f;
#pragma unroll
        for (int e = 0; e < 16; ++e) acc += (float)a[e] * (float)b[e];
#pragma unroll
        for (int off = 1; off < 32; off <<= 1) acc += __shfl_xor(acc, off, 32);
        return acc;
    };

    float num    = sEff * dotWith(row);   // diagonal label = 1.0
    float rowsum = 1.0f;
#pragma unroll
    for (int k = 0; k < KN; ++k) {
        float al = fmaxf(alphas[row * KN + k], 0.0f);
        int   col = inv[nbIds[row * KN + k]];  // ids are permutations: found
        num    += al * sEff * dotWith(col);
        rowsum += al;
    }
    if (lane == 0) rowLoss[row] = lse - num / rowsum;
}

// ---------------------------------------------------------------------------
// Kernel 4: deterministic fixed-order reduction of 2*BN row losses -> scalar
// ---------------------------------------------------------------------------
__global__ void reduce_kernel(const float* __restrict__ rowLoss,
                              float* __restrict__ out) {
    __shared__ float sm[256];
    float acc = 0.0f;
    for (int i = threadIdx.x; i < 2 * BN; i += 256) acc += rowLoss[i];
    sm[threadIdx.x] = acc;
    __syncthreads();
    for (int st = 128; st > 0; st >>= 1) {
        if ((int)threadIdx.x < st) sm[threadIdx.x] += sm[threadIdx.x + st];
        __syncthreads();
    }
    if (threadIdx.x == 0) out[0] = sm[0] / (2.0f * BN);
}

// ---------------------------------------------------------------------------
extern "C" void kernel_launch(void* const* d_in, const int* in_sizes, int n_in,
                              void* d_out, int out_size, void* d_ws, size_t ws_size,
                              hipStream_t stream) {
    const float* img     = (const float*)d_in[0];  // [BN*DN]
    const float* txt     = (const float*)d_in[1];  // [BN*DN]
    const float* scale   = (const float*)d_in[2];  // [1]
    const float* alphas  = (const float*)d_in[3];  // [BN*KN]
    const int*   imgIds  = (const int*)d_in[4];    // [BN]
    const int*   txtIds  = (const int*)d_in[5];    // [BN]
    const int*   nbIds   = (const int*)d_in[6];    // [BN*KN]
    float* out = (float*)d_out;

    // workspace layout (bytes)
    char* ws = (char*)d_ws;
    _Float16* imgH  = (_Float16*)(ws);                                   // 4 MB
    _Float16* txtH  = (_Float16*)(ws + (size_t)BN * DN * 2);             // 4 MB
    int* invImg     = (int*)(ws + (size_t)BN * DN * 4);                  // 16 KB
    int* invTxt     = invImg + BN;                                       // 16 KB
    float* partM0   = (float*)(invTxt + BN);                             // 128 KB
    float* partS0   = partM0 + CSPLIT * BN;
    float* partM1   = partS0 + CSPLIT * BN;
    float* partS1   = partM1 + CSPLIT * BN;
    float* rowLoss  = partS1 + CSPLIT * BN;                              // 32 KB

    // 1) convert to f16 + inverse permutations
    prep_kernel<<<(BN * DN + 255) / 256, 256, 0, stream>>>(
        img, txt, imgIds, txtIds, imgH, txtH, invImg, invTxt);

    // 2) logsumexp of s*(img@txt^T) rows, and of s*(txt@img^T) rows
    dim3 lseGrid(BN / ROWS_PER_BLOCK, CSPLIT);
    lse_kernel<<<lseGrid, 256, LDS_BYTES, stream>>>(imgH, txtH, scale, partM0, partS0);
    lse_kernel<<<lseGrid, 256, LDS_BYTES, stream>>>(txtH, imgH, scale, partM1, partS1);

    // 3) per-row losses (labels_i_t uses text ids; labels_t_i uses image ids)
    row_loss_kernel<<<BN / 8, 256, 0, stream>>>(
        imgH, txtH, scale, alphas, nbIds, invTxt, partM0, partS0, rowLoss);
    row_loss_kernel<<<BN / 8, 256, 0, stream>>>(
        txtH, imgH, scale, alphas, nbIds, invImg, partM1, partS1, rowLoss + BN);

    // 4) deterministic scalar reduction
    reduce_kernel<<<1, 256, 0, stream>>>(rowLoss, out);
}
